// MultiHead_2001454760314
// MI455X (gfx1250) — compile-verified, matrix operand order checked by reading, not run on hardware
//
#include <hip/hip_runtime.h>
#include <hip/hip_bf16.h>
#include <stdint.h>
#include <math.h>

typedef __bf16 bf16_t;
typedef __attribute__((ext_vector_type(16))) __bf16 v16bf;
typedef __attribute__((ext_vector_type(8)))  __bf16 v8bf;
typedef __attribute__((ext_vector_type(8)))  float  v8f;
typedef __attribute__((ext_vector_type(4)))  unsigned int v4u;

#define B_  4
#define L_  2048
#define D_  1024
#define H_  16
#define DH_ 64

// Load one 32-k-slice bf16 WMMA fragment for this lane.
// Per ISA 16-bit A/B layout: lane half selects k-offset; elements are two
// contiguous 8-element (16B) runs: [8*half .. +8) and [16+8*half .. +8).
static __device__ __forceinline__ v16bf ld_frag(const bf16_t* base, int half) {
  union { v16bf v; v8bf h[2]; } u;
  u.h[0] = *(const v8bf*)(base + 8 * half);
  u.h[1] = *(const v8bf*)(base + 16 + 8 * half);
  return u.v;
}

// ---------------------------------------------------------------- converts
__global__ void f32_to_bf16_k(const float* __restrict__ in, bf16_t* __restrict__ out, size_t n) {
  size_t i = (size_t)blockIdx.x * blockDim.x + threadIdx.x;
  size_t stride = (size_t)gridDim.x * blockDim.x;
  for (; i < n; i += stride) out[i] = (bf16_t)in[i];
}

// in: KxN f32 (row-major) -> out: NxK bf16 (row-major, i.e. transposed)
__global__ __launch_bounds__(256)
void transpose_f32_to_bf16_k(const float* __restrict__ in, bf16_t* __restrict__ out, int K, int N) {
  __shared__ float tile[32][33];
  int k0 = blockIdx.x * 32, n0 = blockIdx.y * 32;
  int tx = threadIdx.x & 31, ty = threadIdx.x >> 5;
#pragma unroll
  for (int r = ty; r < 32; r += 8) tile[r][tx] = in[(size_t)(k0 + r) * N + n0 + tx];
  __syncthreads();
#pragma unroll
  for (int r = ty; r < 32; r += 8) out[(size_t)(n0 + r) * K + k0 + tx] = (bf16_t)tile[tx][r];
}

// ---------------------------------------------------------------- GEMM
// C(M,N) = A(M,K) * Bt(N,K)^T, bf16 in, f32 accumulate via WMMA.
// Block: 256 thr = 8 waves, tile 128(M) x 64(N), BK=32. Wave w owns 16 rows.
// Register double-buffered global->LDS pipeline; batched fragment loads so the
// four WMMAs issue back-to-back after a single DS wait.
// EPI 0: scatter QKV per the reference's flat reshape. EPI 1: f32 store.
template<int EPI>
__global__ __launch_bounds__(256)
void gemm_bf16_k(const bf16_t* __restrict__ A, const bf16_t* __restrict__ Bt,
                 int M, int N, int K,
                 bf16_t* __restrict__ Qp, bf16_t* __restrict__ Kp, bf16_t* __restrict__ Vp,
                 float* __restrict__ Outf) {
  __shared__ __align__(16) bf16_t As[128 * 40];  // stride 40 (80B): conflict-free frags
  __shared__ __align__(16) bf16_t Bs[64 * 40];

  const int t = threadIdx.x;
  const int w = t >> 5, lane = t & 31, half = lane >> 4, l16 = lane & 15;
  const int m0 = blockIdx.y * 128, n0 = blockIdx.x * 64;

  // loader geometry (constant per thread)
  const int ar0 = t >> 2,          ac0 = (t & 3) * 8;          // A chunk 0
  const int ar1 = (t + 256) >> 2,  ac1 = ac0;                  // A chunk 1
  const int br  = t >> 2,          bc  = (t & 3) * 8;          // B chunk

  v8f acc[4];
#pragma unroll
  for (int j = 0; j < 4; ++j) acc[j] = (v8f){0.f, 0.f, 0.f, 0.f, 0.f, 0.f, 0.f, 0.f};

  // preload k-tile 0 into registers
  v4u aR0 = *(const v4u*)(A + (size_t)(m0 + ar0) * K + ac0);
  v4u aR1 = *(const v4u*)(A + (size_t)(m0 + ar1) * K + ac1);
  v4u bR  = *(const v4u*)(Bt + (size_t)(n0 + br) * K + bc);

  for (int kk = 0; kk < K; kk += 32) {
    __syncthreads();                       // prior reads of LDS done
    *(v4u*)(As + ar0 * 40 + ac0) = aR0;
    *(v4u*)(As + ar1 * 40 + ac1) = aR1;
    *(v4u*)(Bs + br * 40 + bc)   = bR;
    __syncthreads();

    if (kk + 32 < K) {                     // issue next tile's global loads now;
      int kn = kk + 32;                    // they complete during the WMMAs below
      aR0 = *(const v4u*)(A + (size_t)(m0 + ar0) * K + kn + ac0);
      aR1 = *(const v4u*)(A + (size_t)(m0 + ar1) * K + kn + ac1);
      bR  = *(const v4u*)(Bt + (size_t)(n0 + br) * K + kn + bc);
    }

    v16bf a = ld_frag(As + (w * 16 + l16) * 40, half);
    v16bf bf[4];
#pragma unroll
    for (int j = 0; j < 4; ++j) bf[j] = ld_frag(Bs + (j * 16 + l16) * 40, half);
#pragma unroll
    for (int j = 0; j < 4; ++j)
      acc[j] = __builtin_amdgcn_wmma_f32_16x16x32_bf16(false, a, false, bf[j],
                                                       (short)0, acc[j], false, false);
  }

#pragma unroll
  for (int j = 0; j < 4; ++j) {
#pragma unroll
    for (int e = 0; e < 8; ++e) {
      int n = n0 + j * 16 + l16;
      int m = m0 + w * 16 + e + 8 * half;
      float val = acc[j][e];
      if (EPI == 0) {
        // reference: (x@Wqkv).reshape(B,3,H,L,DH) -- flat reshape of (L,3D)
        unsigned bb = (unsigned)m >> 11;
        unsigned G  = ((unsigned)m & 2047u) * 3072u + (unsigned)n;
        unsigned s  = G >> 21;
        unsigned hh = (G >> 17) & 15u;
        unsigned l2 = (G >> 6) & 2047u;
        unsigned dh = G & 63u;
        size_t idx = (((size_t)bb * H_ + hh) * L_ + l2) * DH_ + dh;
        bf16_t* dst = (s == 0) ? Qp : ((s == 1) ? Kp : Vp);
        dst[idx] = (bf16_t)val;
      } else {
        Outf[(size_t)m * N + n] = val;
      }
    }
  }
}

// ---------------------------------------------------------------- flash attention
// grid: (L/64, H, B); block: 128 thr = 4 waves. Wave w owns q-rows [w*16, w*16+16)
// of a 64-row q tile, so softmax stats live per-wave (shfl within 16-lane halves).
// K/V tiles are register double-buffered from global.
__global__ __launch_bounds__(128)
void flash_attn_k(const bf16_t* __restrict__ Qb, const bf16_t* __restrict__ Kb,
                  const bf16_t* __restrict__ Vb, bf16_t* __restrict__ AO) {
  __shared__ __align__(16) bf16_t Ks[64 * 72];        // [key][dh], stride 72
  __shared__ __align__(16) bf16_t Vst[64 * 72];       // [dh][key] (transposed)
  __shared__ __align__(16) bf16_t Ps[4 * 16 * 72];    // per-wave P staging

  const int t = threadIdx.x;
  const int w = t >> 5, lane = t & 31, half = lane >> 4, l16 = lane & 15;
  const int qt = blockIdx.x, h = blockIdx.y, b = blockIdx.z;
  const size_t bh = (size_t)b * H_ + h;
  const bf16_t* Qbase = Qb + (bh * L_ + (size_t)qt * 64) * DH_;
  const bf16_t* Kbase = Kb + bh * L_ * DH_;
  const bf16_t* Vbase = Vb + bh * L_ * DH_;

  // loader geometry: 4 chunks each of K and V per thread (64x64 bf16 tiles)
  int lrow[4], lcol[4];
#pragma unroll
  for (int i = 0; i < 4; ++i) {
    int c = t + i * 128;
    lrow[i] = c >> 3;
    lcol[i] = (c & 7) * 8;
  }

  // Q fragments kept in registers for the whole pass (K-dim = DH = 64 -> 2 frags)
  const bf16_t* qrow = Qbase + (size_t)(w * 16 + l16) * DH_;
  v16bf qf0 = ld_frag(qrow + 0, half);
  v16bf qf1 = ld_frag(qrow + 32, half);

  v8f o[4];
#pragma unroll
  for (int j = 0; j < 4; ++j) o[j] = (v8f){0.f, 0.f, 0.f, 0.f, 0.f, 0.f, 0.f, 0.f};
  float Mrun[8], Srun[8];
#pragma unroll
  for (int e = 0; e < 8; ++e) { Mrun[e] = -__builtin_inff(); Srun[e] = 0.f; }

  // preload tile 0
  v4u kR[4], vR[4];
#pragma unroll
  for (int i = 0; i < 4; ++i) {
    kR[i] = *(const v4u*)(Kbase + (size_t)lrow[i] * DH_ + lcol[i]);
    vR[i] = *(const v4u*)(Vbase + (size_t)lrow[i] * DH_ + lcol[i]);
  }

  for (int kt = 0; kt < L_ / 64; ++kt) {
    __syncthreads();                       // prior tile's LDS reads done
    // stage K row-major, V transposed, from registers
#pragma unroll
    for (int i = 0; i < 4; ++i) {
      *(v4u*)(Ks + lrow[i] * 72 + lcol[i]) = kR[i];
      union { v4u u; bf16_t hb[8]; } vv;
      vv.u = vR[i];
#pragma unroll
      for (int ii = 0; ii < 8; ++ii) Vst[(lcol[i] + ii) * 72 + lrow[i]] = vv.hb[ii];
    }
    __syncthreads();

    if (kt + 1 < L_ / 64) {                // next tile's global loads fly during compute
      const bf16_t* Kt = Kbase + (size_t)(kt + 1) * 64 * DH_;
      const bf16_t* Vt = Vbase + (size_t)(kt + 1) * 64 * DH_;
#pragma unroll
      for (int i = 0; i < 4; ++i) {
        kR[i] = *(const v4u*)(Kt + (size_t)lrow[i] * DH_ + lcol[i]);
        vR[i] = *(const v4u*)(Vt + (size_t)lrow[i] * DH_ + lcol[i]);
      }
    }

    // S = Q * K^T  (16q x 64k per wave), scaled by DH^-0.5 = 0.125
    v8f s[4];
#pragma unroll
    for (int j = 0; j < 4; ++j) s[j] = (v8f){0.f, 0.f, 0.f, 0.f, 0.f, 0.f, 0.f, 0.f};
#pragma unroll
    for (int kk = 0; kk < 64; kk += 32) {
      v16bf a = (kk == 0) ? qf0 : qf1;
      v16bf bf[4];
#pragma unroll
      for (int j = 0; j < 4; ++j) bf[j] = ld_frag(Ks + (j * 16 + l16) * 72 + kk, half);
#pragma unroll
      for (int j = 0; j < 4; ++j)
        s[j] = __builtin_amdgcn_wmma_f32_16x16x32_bf16(false, a, false, bf[j],
                                                       (short)0, s[j], false, false);
    }
#pragma unroll
    for (int j = 0; j < 4; ++j) s[j] = s[j] * 0.125f;

    // Online softmax: row m = e + 8*half lives on 16 lanes of this half.
    float corr[8];
#pragma unroll
    for (int e = 0; e < 8; ++e) {
      float mx = fmaxf(fmaxf(s[0][e], s[1][e]), fmaxf(s[2][e], s[3][e]));
#pragma unroll
      for (int msk = 1; msk < 16; msk <<= 1) mx = fmaxf(mx, __shfl_xor(mx, msk, 32));
      float mnew = fmaxf(Mrun[e], mx);
      corr[e] = __expf(Mrun[e] - mnew);
      Mrun[e] = mnew;
#pragma unroll
      for (int j = 0; j < 4; ++j) s[j][e] = __expf(s[j][e] - mnew);
      float rs = s[0][e] + s[1][e] + s[2][e] + s[3][e];
#pragma unroll
      for (int msk = 1; msk < 16; msk <<= 1) rs += __shfl_xor(rs, msk, 32);
      Srun[e] = Srun[e] * corr[e] + rs;
#pragma unroll
      for (int j = 0; j < 4; ++j) o[j][e] *= corr[e];
    }

    // P: C-layout -> A-layout via per-wave LDS round trip
    bf16_t* pr = Ps + (size_t)w * 16 * 72;
#pragma unroll
    for (int j = 0; j < 4; ++j)
#pragma unroll
      for (int e = 0; e < 8; ++e)
        pr[(e + 8 * half) * 72 + j * 16 + l16] = (bf16_t)s[j][e];
    __syncthreads();

    // O += P * V
#pragma unroll
    for (int kk = 0; kk < 64; kk += 32) {
      v16bf a = ld_frag(pr + (size_t)l16 * 72 + kk, half);
      v16bf bf[4];
#pragma unroll
      for (int j = 0; j < 4; ++j) bf[j] = ld_frag(Vst + (j * 16 + l16) * 72 + kk, half);
#pragma unroll
      for (int j = 0; j < 4; ++j)
        o[j] = __builtin_amdgcn_wmma_f32_16x16x32_bf16(false, a, false, bf[j],
                                                       (short)0, o[j], false, false);
    }
  }

  // Normalize and store AO[b*L + q][h*64 + dh] (bf16)
  float inv[8];
#pragma unroll
  for (int e = 0; e < 8; ++e) inv[e] = 1.0f / Srun[e];
#pragma unroll
  for (int j = 0; j < 4; ++j)
#pragma unroll
    for (int e = 0; e < 8; ++e) {
      int q = qt * 64 + w * 16 + e + 8 * half;
      int col = h * DH_ + j * 16 + l16;
      AO[((size_t)b * L_ + q) * (H_ * DH_) + col] = (bf16_t)(o[j][e] * inv[e]);
    }
}

// ---------------------------------------------------------------- launch
extern "C" void kernel_launch(void* const* d_in, const int* in_sizes, int n_in,
                              void* d_out, int out_size, void* d_ws, size_t ws_size,
                              hipStream_t stream) {
  (void)in_sizes; (void)n_in; (void)out_size; (void)ws_size;
  const float* xin  = (const float*)d_in[0];  // (B,L,D)
  const float* Wqkv = (const float*)d_in[1];  // (D,3D)
  const float* Wo   = (const float*)d_in[2];  // (D,D)
  float* out = (float*)d_out;                 // (B,L,D) f32

  const size_t M = (size_t)B_ * L_;           // 8192
  char* ws = (char*)d_ws;
  size_t off = 0;
  auto carve = [&](size_t elems) -> bf16_t* {
    bf16_t* p = (bf16_t*)(ws + off);
    off += ((elems * sizeof(bf16_t)) + 255) & ~(size_t)255;
    return p;
  };
  bf16_t* Xb   = carve(M * D_);               // x in bf16
  bf16_t* Wqt  = carve((size_t)3 * D_ * D_);  // W_qkv^T bf16 (3072 x 1024)
  bf16_t* Wot  = carve((size_t)D_ * D_);      // W_o^T bf16
  bf16_t* Qb   = carve(M * D_);               // (B,H,L,DH)
  bf16_t* Kbuf = carve(M * D_);
  bf16_t* Vbuf = carve(M * D_);
  bf16_t* AO   = carve(M * D_);               // attention out (B*L, H*DH)

  f32_to_bf16_k<<<8192, 256, 0, stream>>>(xin, Xb, M * D_);
  transpose_f32_to_bf16_k<<<dim3(D_ / 32, 3 * D_ / 32), 256, 0, stream>>>(Wqkv, Wqt, D_, 3 * D_);
  transpose_f32_to_bf16_k<<<dim3(D_ / 32, D_ / 32), 256, 0, stream>>>(Wo, Wot, D_, D_);

  gemm_bf16_k<0><<<dim3(3 * D_ / 64, M / 128), 256, 0, stream>>>(
      Xb, Wqt, (int)M, 3 * D_, D_, Qb, Kbuf, Vbuf, nullptr);

  flash_attn_k<<<dim3(L_ / 64, H_, B_), 128, 0, stream>>>(Qb, Kbuf, Vbuf, AO);

  gemm_bf16_k<1><<<dim3(D_ / 64, M / 128), 256, 0, stream>>>(
      AO, Wot, (int)M, D_, D_, nullptr, nullptr, nullptr, out);
}
